// FrequencyLayer_36515811950688
// MI455X (gfx1250) — matrix operand order repaired and checked by reading, not compiled, not run on hardware
//
#include <hip/hip_runtime.h>

// ---------------------------------------------------------------------------
// FrequencyLayer: separable 256-pt real DFT, 128x128 low-freq output.
//   real = C^T X^T C - S^T X^T S ;  imag = -(C^T X^T S + S^T X^T C)  per (b,c)
// One workgroup per (b,c) slice; intermediates live in LDS (288 KB of 320 KB).
// bf16 WMMA (v_wmma_f32_16x16x32_bf16) with f32 accumulation.
// DFT tables DMA'd into LDS with the Tensor Data Mover (tensor_load_to_lds).
// ---------------------------------------------------------------------------

typedef __attribute__((ext_vector_type(16))) __bf16 v16bf;
typedef __attribute__((ext_vector_type(8)))  __bf16 v8bf;
typedef __attribute__((ext_vector_type(4)))  __bf16 v4bf;
typedef __attribute__((ext_vector_type(8)))  float  v8f;
typedef unsigned int __attribute__((ext_vector_type(4))) v4u;
typedef int          __attribute__((ext_vector_type(8))) v8i;
typedef int          __attribute__((ext_vector_type(4))) v4i;

#define NDIM   256          // spatial size (H == W)
#define KF     128          // kept frequencies per axis
#define BCN    1024         // B*C
#define IMAG_OFF (static_cast<long>(BCN) * KF * KF)   // 16,777,216 floats

union FragU { v16bf v; v8bf h[2]; };

// A-operand fragment, 16x32 bf16, source row-major [row][k], row stride ld.
// ISA layout: lanes 0-15 row M=lane, K={0..7,16..23}; lanes 16-31 K={8..15,24..31}.
__device__ __forceinline__ v16bf load_a_frag(const __bf16* __restrict__ base,
                                             int row0, int ld, int k0,
                                             int mrow, int hf) {
  FragU u;
  const __bf16* p = base + (row0 + mrow) * ld + k0 + hf * 8;
  u.h[0] = *(const v8bf*)(p);        // K = k0 + hf*8   .. +7
  u.h[1] = *(const v8bf*)(p + 16);   // K = k0+16+hf*8  .. +7
  return u.v;
}

// B-operand fragment, 32x16 bf16, source stored TRANSPOSED [n][k], stride ldT.
// ISA layout: lanes 0-15 col N=lane, K={0..7,16..23}; lanes 16-31 K={8..15,24..31}.
__device__ __forceinline__ v16bf load_b_frag(const __bf16* __restrict__ baseT,
                                             int n0, int ldT, int k0,
                                             int ncol, int hf) {
  FragU u;
  const __bf16* p = baseT + (n0 + ncol) * ldT + k0 + hf * 8;
  u.h[0] = *(const v8bf*)(p);        // K = k0 + hf*8   .. +7
  u.h[1] = *(const v8bf*)(p + 16);   // K = k0+16+hf*8  .. +7
  return u.v;
}

__device__ __forceinline__ v8f wmma_bf16(v16bf a, v16bf b, v8f c) {
  return __builtin_amdgcn_wmma_f32_16x16x32_bf16(false, a, false, b,
                                                 (short)0, c, false, false);
}

// Tensor Data Mover: DMA a contiguous 2D bf16 tile (dim0 fast, dim1 rows)
// from global memory into LDS at byte offset lds_off. D# per CDNA5 ISA §8.
// This toolchain exposes the 6-arg builtin: (g0 v4u, g1 v8i, g2 v4i, g3 v4i,
// g4 v8i, i32 cpol).
__device__ __forceinline__ void tdm_load_2d_bf16(const __bf16* gsrc,
                                                 unsigned int lds_off,
                                                 int dim0, int dim1) {
  unsigned long long ga = (unsigned long long)(__SIZE_TYPE__)gsrc;
  v4u g0;
  g0[0] = 1u;                                    // count=1, is_restore=0
  g0[1] = lds_off;                               // lds_addr (bytes)
  g0[2] = (unsigned int)(ga & 0xffffffffu);      // global_addr[31:0]
  g0[3] = (unsigned int)((ga >> 32) & 0x01ffffffu) | (2u << 30); // type=2
  v8i g1;
  g1[0] = (1 << 16);                             // data_size = 2 bytes
  g1[1] = (int)((unsigned)dim0 << 16);           // tensor_dim0 (low16 @bit48)
  g1[2] = (int)((unsigned)dim1 << 16);           // tensor_dim1 (low16 @bit80)
  g1[3] = (int)((unsigned)dim0 << 16);           // tile_dim0   (@bit112)
  g1[4] = dim1;                                  // tile_dim1   (@bit128)
  g1[5] = dim0;                                  // tensor_dim0_stride (@bit160)
  g1[6] = 0;
  g1[7] = 0;
  v4i z4 = {0, 0, 0, 0};
  v8i z8 = {0, 0, 0, 0, 0, 0, 0, 0};
  __builtin_amdgcn_tensor_load_to_lds(g0, g1, z4, z4, z8, 0);
}

// --------------------------- table init kernel -----------------------------
// Builds transposed bf16 DFT tables in workspace:  gC[u][j] = cos(2*pi*j*u/256)
__global__ void dft_tables_kernel(__bf16* __restrict__ gC,
                                  __bf16* __restrict__ gS) {
  int t = blockIdx.x * blockDim.x + threadIdx.x;   // 0..32767
  int u = t >> 8;
  int j = t & 255;
  int p = (j * u) & 255;                           // exact mod-N phase reduction
  float ang = (float)p * (6.283185307179586f / 256.0f);
  float s, c;
  __sincosf(ang, &s, &c);
  gC[t] = (__bf16)c;
  gS[t] = (__bf16)s;
}

// ----------------------------- main kernel ---------------------------------
__global__ __launch_bounds__(256)
void freq_dft_kernel(const float* __restrict__ x,
                     const __bf16* __restrict__ gC,
                     const __bf16* __restrict__ gS,
                     float* __restrict__ out) {
  extern __shared__ char smem[];
  __bf16* sCt = (__bf16*)smem;          // [128][256] cos, transposed (u,j)
  __bf16* sSt = sCt + 128 * 256;        // [128][256] sin, transposed
  __bf16* sA  = sSt + 128 * 256;        // [128][256] stage-1 A, stored (u,i)
  __bf16* sB  = sA  + 128 * 256;        // [128][256] stage-1 B, stored (u,i)
  __bf16* sX  = sB  + 128 * 256;        // [ 64][256] X chunk (i,j)
  // total: 4*64KB + 32KB = 288 KB of the WGP's 320 KB LDS

  const int tid  = threadIdx.x;
  const int lane = tid & 31;
  const int wave = tid >> 5;
  const int hf   = lane >> 4;       // lane half (WMMA striping)
  const int mrow = lane & 15;       // M row / N col within tile
  const int bc   = blockIdx.x;      // (b*64 + c)
  const long xbase = (long)bc * (NDIM * NDIM);

  // --- DMA the DFT tables into LDS via the Tensor Data Mover (wave 0) ---
  if (wave == 0) {
    unsigned int offC = (unsigned int)((const char*)sCt - smem);  // 0
    unsigned int offS = (unsigned int)((const char*)sSt - smem);  // 65536
    tdm_load_2d_bf16(gC, offC, 256, 128);
    tdm_load_2d_bf16(gS, offS, 256, 128);
    __builtin_amdgcn_s_wait_tensorcnt(0);
  }

  // ======================= stage 1:  A = X*C, B = X*S ======================
  // Each wave owns a 2x2 macro-tile (2 i-tiles x 2 u-tiles): 6 fragment
  // loads feed 8 WMMAs per K-step (1.5 ds_load_b128 per WMMA).
  const int iM = (wave >> 2) * 32;    // i macro within chunk: 0 / 32
  const int uM = (wave & 3) * 32;     // u macro: 0 / 32 / 64 / 96

  for (int chunk = 0; chunk < 4; ++chunk) {
    // load 64 rows of X, convert fp32 -> bf16 into sX
    {
      const float4* src = (const float4*)(x + xbase + (long)chunk * 64 * NDIM);
      v4bf* dst = (v4bf*)sX;
#pragma unroll
      for (int q = 0; q < 16; ++q) {
        int idx = tid + 256 * q;    // 4096 float4 per chunk
        float4 f = src[idx];
        v4bf h;
        h[0] = (__bf16)f.x; h[1] = (__bf16)f.y;
        h[2] = (__bf16)f.z; h[3] = (__bf16)f.w;
        dst[idx] = h;
      }
    }
    // prefetch next chunk while we compute on this one (global_prefetch_b8)
    if (chunk < 3) {
      const char* nxt = (const char*)(x + xbase + (long)(chunk + 1) * 64 * NDIM);
      __builtin_prefetch(nxt + (size_t)tid * 128, 0, 1);
      __builtin_prefetch(nxt + (size_t)(tid + 256) * 128, 0, 1);
    }
    __syncthreads();   // (first pass also publishes the TDM'd tables)

    v8f accA[2][2] = {};   // [i-tile][u-tile]  X*C
    v8f accB[2][2] = {};   //                   X*S
#pragma unroll
    for (int kk = 0; kk < 8; ++kk) {
      int k0 = kk * 32;
      v16bf af0 = load_a_frag(sX,  iM,      256, k0, mrow, hf);
      v16bf af1 = load_a_frag(sX,  iM + 16, 256, k0, mrow, hf);
      v16bf bc0 = load_b_frag(sCt, uM,      256, k0, mrow, hf);
      v16bf bc1 = load_b_frag(sCt, uM + 16, 256, k0, mrow, hf);
      v16bf bs0 = load_b_frag(sSt, uM,      256, k0, mrow, hf);
      v16bf bs1 = load_b_frag(sSt, uM + 16, 256, k0, mrow, hf);
      accA[0][0] = wmma_bf16(af0, bc0, accA[0][0]);
      accA[0][1] = wmma_bf16(af0, bc1, accA[0][1]);
      accA[1][0] = wmma_bf16(af1, bc0, accA[1][0]);
      accA[1][1] = wmma_bf16(af1, bc1, accA[1][1]);
      accB[0][0] = wmma_bf16(af0, bs0, accB[0][0]);
      accB[0][1] = wmma_bf16(af0, bs1, accB[0][1]);
      accB[1][0] = wmma_bf16(af1, bs0, accB[1][0]);
      accB[1][1] = wmma_bf16(af1, bs1, accB[1][1]);
    }
    // store D fragments TRANSPOSED (u,i) as bf16; per lane the 8 rows
    // M = r + 8*hf are contiguous in i -> one 16B ds_store per tile
#pragma unroll
    for (int di = 0; di < 2; ++di) {
#pragma unroll
      for (int du = 0; du < 2; ++du) {
        int iG  = chunk * 64 + iM + 16 * di + 8 * hf;
        int row = uM + 16 * du + mrow;
        v8bf ha, hb;
#pragma unroll
        for (int r = 0; r < 8; ++r) {
          ha[r] = (__bf16)accA[di][du][r];
          hb[r] = (__bf16)accB[di][du][r];
        }
        *(v8bf*)(sA + row * 256 + iG) = ha;
        *(v8bf*)(sB + row * 256 + iG) = hb;
      }
    }
    __syncthreads();   // protect sX reuse; final one also publishes sA/sB
  }

  // ============ stage 2: real = A^T C - B^T S, imag = -(B^T C + A^T S) =====
  // Each wave owns one u-tile row; 2 v-tiles blocked per iteration:
  // 6 fragment loads feed 8 WMMAs per K-step.
  const int u0 = wave * 16;
  for (int vt = 0; vt < 4; ++vt) {
    const int v0 = vt * 32;
    v8f accAC[2] = {}, accBS[2] = {}, accBC[2] = {}, accAS[2] = {};
#pragma unroll
    for (int kk = 0; kk < 8; ++kk) {
      int k0 = kk * 32;
      v16bf aA  = load_a_frag(sA,  u0,      256, k0, mrow, hf);
      v16bf aB  = load_a_frag(sB,  u0,      256, k0, mrow, hf);
      v16bf bC0 = load_b_frag(sCt, v0,      256, k0, mrow, hf);
      v16bf bC1 = load_b_frag(sCt, v0 + 16, 256, k0, mrow, hf);
      v16bf bS0 = load_b_frag(sSt, v0,      256, k0, mrow, hf);
      v16bf bS1 = load_b_frag(sSt, v0 + 16, 256, k0, mrow, hf);
      accAC[0] = wmma_bf16(aA, bC0, accAC[0]);
      accAC[1] = wmma_bf16(aA, bC1, accAC[1]);
      accBS[0] = wmma_bf16(aB, bS0, accBS[0]);
      accBS[1] = wmma_bf16(aB, bS1, accBS[1]);
      accBC[0] = wmma_bf16(aB, bC0, accBC[0]);
      accBC[1] = wmma_bf16(aB, bC1, accBC[1]);
      accAS[0] = wmma_bf16(aA, bS0, accAS[0]);
      accAS[1] = wmma_bf16(aA, bS1, accAS[1]);
    }
    // D layout: M = r + 8*hf, N = mrow; rows go out with stride KF (coalesced
    // 64B per half-wave). Outputs concatenated: [real | imag].
    const int Mb = u0 + 8 * hf;
#pragma unroll
    for (int dv = 0; dv < 2; ++dv) {
      const int Nc = v0 + 16 * dv + mrow;
      float* pr = out + ((long)bc * KF + Mb) * KF + Nc;
      float* pi = pr + IMAG_OFF;
#pragma unroll
      for (int r = 0; r < 8; ++r) {
        pr[(long)r * KF] = accAC[dv][r] - accBS[dv][r];
        pi[(long)r * KF] = -(accBC[dv][r] + accAS[dv][r]);
      }
    }
  }
}

// ---------------------------------------------------------------------------
extern "C" void kernel_launch(void* const* d_in, const int* in_sizes, int n_in,
                              void* d_out, int out_size, void* d_ws, size_t ws_size,
                              hipStream_t stream) {
  const float* x   = (const float*)d_in[0];
  float*       out = (float*)d_out;
  __bf16* gC = (__bf16*)d_ws;            // 128*256 bf16 = 64 KB
  __bf16* gS = gC + 128 * 256;           // next 64 KB

  (void)in_sizes; (void)n_in; (void)out_size; (void)ws_size;

  // allow 288 KB dynamic LDS (WGP supports 320 KB); host-side, capture-safe
  size_t shmem = (size_t)(4 * 128 * 256 + 64 * 256) * sizeof(__bf16); // 294912
  (void)hipFuncSetAttribute((const void*)freq_dft_kernel,
                            hipFuncAttributeMaxDynamicSharedMemorySize,
                            (int)shmem);

  dft_tables_kernel<<<128, 256, 0, stream>>>(gC, gS);
  freq_dft_kernel<<<BCN, 256, shmem, stream>>>(x, gC, gS, out);
}
